// SelfAttn_79920751444245
// MI455X (gfx1250) — compile-verified, matrix operand order checked by reading, not run on hardware
//
#include <hip/hip_runtime.h>
#include <math.h>

// Problem constants (from reference)
#define NAY 15
#define NEN 16
#define DD  128
#define AA  64
#define OTH 160
#define ROWLEN 4128   // NAY*DD + NEN*DD + OTH
#define BT  16        // batch rows per workgroup
#define WPAD 130      // f16 row pitch for staged weights (65-dword stride -> conflict-free)

typedef __attribute__((ext_vector_type(16))) _Float16 v16h;
typedef __attribute__((ext_vector_type(8)))  float    v8f;

struct Params { const float* p[35]; };

__device__ __forceinline__ v8f wmma16(v16h a, v16h b, v8f c) {
  // D = A(16x32) * B(32x16) + C, f16 in / f32 out
  return __builtin_amdgcn_wmma_f32_16x16x32_f16(
      /*neg_a=*/false, a, /*neg_b=*/false, b,
      /*c_mod=*/(short)0, c, /*reuse_a=*/false, /*reuse_b=*/false);
}

// A/B fragment loader. 16-bit A 16x32 layout (ISA 7.12.2):
// lane holds row (lane&15); half h = lane>>4.
// frag elems 0..7  -> K = kc*32 + 8h + i
// frag elems 8..15 -> K = kc*32 + 16 + 8h + (i-8)
__device__ __forceinline__ v16h fragf32(const float* row, int kc, int h) {
  const float* p = row + kc * 32 + 8 * h;
  v16h f;
#pragma unroll
  for (int i = 0; i < 8; ++i) f[i] = (_Float16)p[i];
#pragma unroll
  for (int i = 0; i < 8; ++i) f[8 + i] = (_Float16)p[16 + i];
  return f;
}

__device__ __forceinline__ v16h fragf16(const _Float16* row, int kc, int h) {
  const _Float16* p = row + kc * 32 + 8 * h;
  v16h f;
#pragma unroll
  for (int i = 0; i < 8; ++i) f[i] = p[i];
#pragma unroll
  for (int i = 0; i < 8; ++i) f[8 + i] = p[16 + i];
  return f;
}

__device__ __forceinline__ float redsum16(float x) {
  x += __shfl_xor(x, 1, 16);
  x += __shfl_xor(x, 2, 16);
  x += __shfl_xor(x, 4, 16);
  x += __shfl_xor(x, 8, 16);
  return x;
}

// 16-row projection, A rows f32 (global), B = staged f16 weights in LDS.
template <int NKC>
__device__ __forceinline__ void projL(const float* myrow,
                                      const _Float16 (*Wl)[WPAD],
                                      const float* bias, v8f acc[4], int lane) {
  const int h = lane >> 4, n = lane & 15;
#pragma unroll
  for (int ct = 0; ct < 4; ++ct) {
    float b = bias[ct * 16 + n];
#pragma unroll
    for (int j = 0; j < 8; ++j) acc[ct][j] = b;
  }
#pragma unroll
  for (int kc = 0; kc < NKC; ++kc) {
    v16h a = fragf32(myrow, kc, h);
#pragma unroll
    for (int ct = 0; ct < 4; ++ct)
      acc[ct] = wmma16(a, fragf16(&Wl[ct * 16 + n][0], kc, h), acc[ct]);
  }
}

// K and V projections sharing one A-fragment load per k-chunk (B from LDS f16)
__device__ __forceinline__ void proj_dualL(const float* myrow,
                                           const _Float16 (*Wk)[WPAD],
                                           const float* bk,
                                           const _Float16 (*Wv)[WPAD],
                                           const float* bv, v8f accK[4],
                                           v8f accV[4], int lane) {
  const int h = lane >> 4, n = lane & 15;
#pragma unroll
  for (int ct = 0; ct < 4; ++ct) {
    float b0 = bk[ct * 16 + n], b1 = bv[ct * 16 + n];
#pragma unroll
    for (int j = 0; j < 8; ++j) { accK[ct][j] = b0; accV[ct][j] = b1; }
  }
#pragma unroll
  for (int kc = 0; kc < 4; ++kc) {
    v16h a = fragf32(myrow, kc, h);
#pragma unroll
    for (int ct = 0; ct < 4; ++ct) {
      accK[ct] = wmma16(a, fragf16(&Wk[ct * 16 + n][0], kc, h), accK[ct]);
      accV[ct] = wmma16(a, fragf16(&Wv[ct * 16 + n][0], kc, h), accV[ct]);
    }
  }
}

// 16-row projection with A f32 rows and B f32 global rows (phase-2 GEMMs).
template <int NKC, int LDW>
__device__ __forceinline__ void projT(const float* myrow, const float* W,
                                      const float* bias, v8f acc[4], int lane) {
  const int h = lane >> 4, n = lane & 15;
#pragma unroll
  for (int ct = 0; ct < 4; ++ct) {
    float b = bias[ct * 16 + n];
#pragma unroll
    for (int j = 0; j < 8; ++j) acc[ct][j] = b;
  }
#pragma unroll
  for (int kc = 0; kc < NKC; ++kc) {
    v16h a = fragf32(myrow, kc, h);
#pragma unroll
    for (int ct = 0; ct < 4; ++ct)
      acc[ct] = wmma16(a, fragf32(W + (ct * 16 + n) * LDW, kc, h), acc[ct]);
  }
}

// A rows f16 (LDS), B f32 global (fc2).
template <int NKC, int LDW>
__device__ __forceinline__ void projT16(const _Float16* myrow, const float* W,
                                        const float* bias, v8f acc[4], int lane) {
  const int h = lane >> 4, n = lane & 15;
#pragma unroll
  for (int ct = 0; ct < 4; ++ct) {
    float b = bias[ct * 16 + n];
#pragma unroll
    for (int j = 0; j < 8; ++j) acc[ct][j] = b;
  }
#pragma unroll
  for (int kc = 0; kc < NKC; ++kc) {
    v16h a = fragf16(myrow, kc, h);
#pragma unroll
    for (int ct = 0; ct < 4; ++ct)
      acc[ct] = wmma16(a, fragf32(W + (ct * 16 + n) * LDW, kc, h), acc[ct]);
  }
}

// Row-wise LayerNorm over 64 cols directly on the WMMA accumulator layout.
// Row M = j + 8h lives in half-wave h; cols N = ct*16 + (lane&15).
__device__ __forceinline__ void ln_rows(v8f acc[4], const float* g,
                                        const float* b, int lane) {
  const int n = lane & 15;
  float gv[4], bv[4];
#pragma unroll
  for (int ct = 0; ct < 4; ++ct) { gv[ct] = g[ct * 16 + n]; bv[ct] = b[ct * 16 + n]; }
#pragma unroll
  for (int j = 0; j < 8; ++j) {
    float s = 0.f, q = 0.f;
#pragma unroll
    for (int ct = 0; ct < 4; ++ct) { float x = acc[ct][j]; s += x; q += x * x; }
    s = redsum16(s);
    q = redsum16(q);
    float m = s * (1.f / 64.f);
    float var = q * (1.f / 64.f) - m * m;
    float r = rsqrtf(var + 1e-6f);
#pragma unroll
    for (int ct = 0; ct < 4; ++ct)
      acc[ct][j] = (acc[ct][j] - m) * r * gv[ct] + bv[ct];
  }
}

__device__ __forceinline__ void store_rows(float* __restrict__ out, int b0,
                                           int nb, int coloff, v8f acc[4],
                                           int lane) {
  const int h = lane >> 4, n = lane & 15;
#pragma unroll
  for (int j = 0; j < 8; ++j) {
    int gb = b0 + j + 8 * h;
    if (gb < nb) {
#pragma unroll
      for (int ct = 0; ct < 4; ++ct)
        out[(size_t)gb * 192 + coloff + ct * 16 + n] = acc[ct][j];
    }
  }
}

__global__ __launch_bounds__(256) void attn_fused(Params prm,
                                                  float* __restrict__ out,
                                                  int nb) {
  const int tid = threadIdx.x;
  const int lane = tid & 31;
  const int wid = tid >> 5;
  const int h = lane >> 4;
  const int n15 = lane & 15;
  const int b0 = blockIdx.x * BT;
  const float* inp = prm.p[0];

  // Staged f16 weights: [set*3 + {wq,wk,wv}][row][col], 130-half pitch.
  __shared__ _Float16 s_w[6][64][WPAD];        // ~100 KB
  __shared__ float    s_o[2][16][64];          // attention outputs (a/e sets)
  __shared__ _Float16 s_m[2][16][64];          // relu(fc1) intermediates

  // ---- Phase 0: convert the 6 hot weight matrices to f16 in LDS, once ----
  {
    const int midx[6] = {1, 3, 5, 17, 19, 21};  // a_wq,a_wk,a_wv,e_wq,e_wk,e_wv
#pragma unroll 1
    for (int m = 0; m < 6; ++m) {
      const float* Wsrc = prm.p[midx[m]];
      for (int t = tid * 4; t < AA * DD; t += 256 * 4) {
        const float4 x = *(const float4*)(Wsrc + t);
        const int r = t >> 7, c = t & 127;
        s_w[m][r][c + 0] = (_Float16)x.x;
        s_w[m][r][c + 1] = (_Float16)x.y;
        s_w[m][r][c + 2] = (_Float16)x.z;
        s_w[m][r][c + 3] = (_Float16)x.w;
      }
    }
  }
  __syncthreads();

  // ---- Phase 1: each wave owns 2 batch rows; both attention heads ----
  // Both loops kept rolled to hold one attention body's register footprint
  // (<256 VGPRs -> no vgpr-msb churn, more resident waves for latency hiding).
#pragma unroll 1
  for (int lb2 = 0; lb2 < 2; ++lb2) {
    const int lb = wid * 2 + lb2;
    int gb = b0 + lb;
    if (gb >= nb) gb = nb - 1;
    const float* xrow = inp + (size_t)gb * ROWLEN;
    const float* ownp = xrow + (ROWLEN - DD);
    __builtin_prefetch(xrow, 0, 1);
    __builtin_prefetch(ownp, 0, 1);

#pragma unroll 1
    for (int set = 0; set < 2; ++set) {
      // param bundle: [wq,bq,wk,bk,wv,bv, lnq_g,lnq_b,lnk_g,lnk_b,lnv_g,lnv_b,
      //                fc1_w,fc1_b,fc2_w,fc2_b]
      const float* const* W = prm.p + (set ? 17 : 1);
      const float* kvbase = set ? (xrow + NAY * DD) : xrow;
      int r = n15;
      if (set == 0 && r == NAY) r = 0;  // ally pad row -> duplicate row 0
      const float* myrow = kvbase + r * DD;

      // q = LN(own @ wq^T + bq), broadcast over all 16 tile rows
      v8f accQ[4];
      projL<4>(ownp, s_w[set * 3 + 0], W[1], accQ, lane);
      ln_rows(accQ, W[6], W[7], lane);

      // k,v = LN(kv @ w^T + b)
      v8f accK[4], accV[4];
      proj_dualL(myrow, s_w[set * 3 + 1], W[3], s_w[set * 3 + 2], W[5],
                 accK, accV, lane);
      ln_rows(accK, W[8], W[9], lane);
      ln_rows(accV, W[10], W[11], lane);

      // scores[M] = (q . k_M) / sqrt(64): reduce over N within half-wave
      float sc[8];
#pragma unroll
      for (int j = 0; j < 8; ++j) {
        float p = 0.f;
#pragma unroll
        for (int ct = 0; ct < 4; ++ct) p += accK[ct][j] * accQ[ct][j];
        sc[j] = redsum16(p) * 0.125f;
      }
      if (set == 0 && h == 1) sc[7] = -3.0e38f;  // mask pad row 15 (ally)

      // softmax across the 16 rows (8 per half-wave + cross-half combine)
      float mx = sc[0];
#pragma unroll
      for (int j = 1; j < 8; ++j) mx = fmaxf(mx, sc[j]);
      mx = fmaxf(mx, __shfl_xor(mx, 16, 32));
      float ex[8];
      float den = 0.f;
#pragma unroll
      for (int j = 0; j < 8; ++j) { ex[j] = __expf(sc[j] - mx); den += ex[j]; }
      den += __shfl_xor(den, 16, 32);
      const float inv = 1.f / den;

      // o[N] = sum_M attn[M] * V[M][N]
#pragma unroll
      for (int ct = 0; ct < 4; ++ct) {
        float oc = 0.f;
#pragma unroll
        for (int j = 0; j < 8; ++j) oc += ex[j] * accV[ct][j];
        oc += __shfl_xor(oc, 16, 32);
        if (h == 0) s_o[set][lb][ct * 16 + n15] = oc * inv;
      }
    }
  }
  __syncthreads();

  // ---- Phase 2a: fc1 (+relu) for both sets, and the "other" linear ----
  if (wid < 2) {
    const float* const* W = prm.p + (wid ? 17 : 1);
    v8f acc[4];
    projT<2, AA>(&s_o[wid][n15][0], W[12], W[13], acc, lane);
#pragma unroll
    for (int j = 0; j < 8; ++j) {
      const int M = j + 8 * h;
#pragma unroll
      for (int ct = 0; ct < 4; ++ct)
        s_m[wid][M][ct * 16 + n15] = (_Float16)fmaxf(acc[ct][j], 0.f);
    }
  } else if (wid == 2) {
    int gb = b0 + n15;
    if (gb >= nb) gb = nb - 1;
    const float* myrow = inp + (size_t)gb * ROWLEN + (NAY * DD + NEN * DD);
    v8f acc[4];
    projT<5, OTH>(myrow, prm.p[33], prm.p[34], acc, lane);
    store_rows(out, b0, nb, 128, acc, lane);
  }
  __syncthreads();

  // ---- Phase 2b: fc2, write ao / eo ----
  if (wid < 2) {
    const float* const* W = prm.p + (wid ? 17 : 1);
    v8f acc[4];
    projT16<2, AA>(&s_m[wid][n15][0], W[14], W[15], acc, lane);
    store_rows(out, b0, nb, wid * 64, acc, lane);
  }
}

extern "C" void kernel_launch(void* const* d_in, const int* in_sizes, int n_in,
                              void* d_out, int out_size, void* d_ws,
                              size_t ws_size, hipStream_t stream) {
  (void)d_ws; (void)ws_size; (void)out_size;
  Params prm;
  for (int i = 0; i < 35 && i < n_in; ++i) prm.p[i] = (const float*)d_in[i];
  const int nb = in_sizes[0] / ROWLEN;
  const int blocks = (nb + BT - 1) / BT;
  attn_fused<<<dim3(blocks), dim3(256), 0, stream>>>(prm, (float*)d_out, nb);
}
